// AttentionBlock_45792941310249
// MI455X (gfx1250) — compile-verified, matrix operand order checked by reading, not run on hardware
//
#include <hip/hip_runtime.h>

typedef __attribute__((ext_vector_type(16))) __bf16 v16bf;
typedef __attribute__((ext_vector_type(8)))  float  v8f;
typedef unsigned short u16;
typedef unsigned int   u32;

#define Bc   4
#define NQc  1024
#define NKc  1024
#define DOc  1024
#define Hc   16
#define HDc  64
#define HIDc 2048

union Frag16 { v16bf v; uint4 q[2]; };
union Acc8   { v8f   v; float f[8]; };

__device__ __forceinline__ u16 f2bf(float f) {
  union { float f; u32 u; } c; c.f = f;
  u32 r = c.u + 0x7FFFu + ((c.u >> 16) & 1u);
  return (u16)(r >> 16);
}

// ---------------------------------------------------------------------------
// Elementwise DyT: out = bf16(gamma[col] * tanh(alpha * x))
// ---------------------------------------------------------------------------
__global__ void dyt_cast_kernel(const float* __restrict__ x,
                                const float* __restrict__ alpha_p,
                                const float* __restrict__ gamma,
                                u16* __restrict__ out, int D, long total) {
  long idx = (long)blockIdx.x * blockDim.x + threadIdx.x;
  if (idx >= total) return;
  float a = alpha_p[0];
  int col = (int)(idx % D);
  out[idx] = f2bf(gamma[col] * tanhf(a * x[idx]));
}

// ---------------------------------------------------------------------------
// Weight transpose + cast: W[R,C] f32 -> Wt[C,R] bf16
// ---------------------------------------------------------------------------
__global__ void transpose_cast_kernel(const float* __restrict__ W,
                                      u16* __restrict__ Wt, int R, int C) {
  __shared__ u16 tile[32][33];
  int c0 = blockIdx.x * 32, r0 = blockIdx.y * 32;
  for (int i = threadIdx.y; i < 32; i += 8)
    tile[i][threadIdx.x] = f2bf(W[(long)(r0 + i) * C + c0 + threadIdx.x]);
  __syncthreads();
  for (int i = threadIdx.y; i < 32; i += 8)
    Wt[(long)(c0 + i) * R + r0 + threadIdx.x] = tile[threadIdx.x][i];
}

// ---------------------------------------------------------------------------
// WMMA GEMM: C[M,N] = A[M,K](bf16, row-major) @ Bt[N,K](bf16)^T + epilogue
// Block tile 128(M) x 128(N); 8 waves as 4(M) x 2(N); wave tile 32 x 64.
// K stepped by 32; B tile double-buffered in LDS (one barrier per step);
// 8 WMMAs per wave per K-step (B frag reused x2, A frag reused x4).
// MODE 0: scatter to per-head Q layout   [B,H,NQ,64] bf16
// MODE 1: split KV -> K [B,H,NK,64], V^T [B,H,64,NK] bf16
// MODE 2: +bias; store f32 x; store bf16 dyt(x) (gamma2,alpha2)
// MODE 3: +bias; exact GELU; store bf16
// MODE 4: +bias; +resid f32; store f32 (final output)
// ---------------------------------------------------------------------------
template<int MODE>
__global__ __launch_bounds__(256) void gemm_wmma_kernel(
    const u16* __restrict__ A, const u16* __restrict__ Bt,
    int M, int N, int K,
    const float* __restrict__ bias,
    const float* __restrict__ gamma,
    const float* __restrict__ alpha_p,
    const float* __restrict__ resid,
    u16* __restrict__ outA, u16* __restrict__ outB,
    float* __restrict__ outF) {
  __shared__ __align__(16) u16 ldsB[2][128 * 32];
  const int tid   = threadIdx.x;
  const int wave  = tid >> 5, lane = tid & 31;
  const int waveM = wave & 3, waveN = wave >> 2;
  const int m0    = blockIdx.x * 128 + waveM * 32;
  const int nblk  = blockIdx.y * 128;
  const int nw0   = waveN * 64;            // wave's N offset within block tile
  const int lrow  = lane & 15;
  const int klo   = (lane < 16) ? 0 : 8;   // 16-bit A/B WMMA lane K pattern

  Acc8 acc[2][4];
  #pragma unroll
  for (int mt = 0; mt < 2; ++mt)
    #pragma unroll
    for (int nt = 0; nt < 4; ++nt)
      #pragma unroll
      for (int j = 0; j < 8; ++j) acc[mt][nt].f[j] = 0.0f;

  const u16* aRow0 = A + (long)(m0 + lrow) * K;
  const u16* aRow1 = A + (long)(m0 + 16 + lrow) * K;

  // B-tile staging: 128 rows x 32 k = 512 uint4 chunks, 2 per thread
  const int r0 = (tid + 0)   >> 2, c0_ = (tid + 0)   & 3;
  const int r1 = (tid + 256) >> 2, c1_ = (tid + 256) & 3;
  const u16* btBase = Bt + (long)nblk * K;

  // prologue: fetch first tile into registers
  uint4 pre0 = *(const uint4*)(btBase + (long)r0 * K + c0_ * 8);
  uint4 pre1 = *(const uint4*)(btBase + (long)r1 * K + c1_ * 8);

  for (int k0 = 0; k0 < K; k0 += 32) {
    const int buf = (k0 >> 5) & 1;
    *(uint4*)&ldsB[buf][r0 * 32 + c0_ * 8] = pre0;
    *(uint4*)&ldsB[buf][r1 * 32 + c1_ * 8] = pre1;
    __syncthreads();

    if (k0 + 32 < K) {  // overlap next tile's global loads with compute
      pre0 = *(const uint4*)(btBase + (long)r0 * K + k0 + 32 + c0_ * 8);
      pre1 = *(const uint4*)(btBase + (long)r1 * K + k0 + 32 + c1_ * 8);
      __builtin_prefetch(aRow0 + k0 + 32, 0, 1);
    }

    Frag16 af[2];
    af[0].q[0] = *(const uint4*)(aRow0 + k0 + klo);
    af[0].q[1] = *(const uint4*)(aRow0 + k0 + 16 + klo);
    af[1].q[0] = *(const uint4*)(aRow1 + k0 + klo);
    af[1].q[1] = *(const uint4*)(aRow1 + k0 + 16 + klo);

    #pragma unroll
    for (int nt = 0; nt < 4; ++nt) {
      Frag16 bfr;
      const u16* bp = &ldsB[buf][(nw0 + nt * 16 + lrow) * 32];
      bfr.q[0] = *(const uint4*)(bp + klo);
      bfr.q[1] = *(const uint4*)(bp + 16 + klo);
      #pragma unroll
      for (int mt = 0; mt < 2; ++mt)
        acc[mt][nt].v = __builtin_amdgcn_wmma_f32_16x16x32_bf16(
            false, af[mt].v, false, bfr.v, (short)0, acc[mt][nt].v, false, false);
    }
    // double buffer: next store targets the other half -> no trailing barrier
  }

  const int mhalf = (lane >> 4) << 3;   // C layout: lanes>=16 hold M+8
  float a2 = 0.0f;
  if (MODE == 2) a2 = alpha_p[0];

  #pragma unroll
  for (int mt = 0; mt < 2; ++mt) {
    #pragma unroll
    for (int nt = 0; nt < 4; ++nt) {
      int gn = nblk + nw0 + nt * 16 + lrow;
      #pragma unroll
      for (int r = 0; r < 8; ++r) {
        int gm = m0 + mt * 16 + mhalf + r;
        float val = acc[mt][nt].f[r];
        if (MODE == 0) {
          int b = gm >> 10, i = gm & 1023;
          int h = gn >> 6,  d = gn & 63;
          outA[((((long)b * Hc + h) * NQc) + i) * HDc + d] = f2bf(val);
        } else if (MODE == 1) {
          int b = gm >> 10, j = gm & 1023;
          int c = gn >> 10, h = (gn >> 6) & 15, d = gn & 63;
          if (c == 0)
            outA[((((long)b * Hc + h) * NKc) + j) * HDc + d] = f2bf(val);
          else
            outB[((((long)b * Hc + h) * HDc) + d) * NKc + j] = f2bf(val);
        } else if (MODE == 2) {
          val += bias[gn];
          outF[(long)gm * N + gn] = val;
          outA[(long)gm * N + gn] = f2bf(gamma[gn] * tanhf(a2 * val));
        } else if (MODE == 3) {
          val += bias[gn];
          float g = 0.5f * val * (1.0f + erff(val * 0.70710678f));
          outA[(long)gm * N + gn] = f2bf(g);
        } else {
          val += bias[gn];
          outF[(long)gm * N + gn] = resid[(long)gm * N + gn] + val;
        }
      }
    }
  }
}

// ---------------------------------------------------------------------------
// Flash attention: block = (b, h, 64 queries); 4 waves, 16 queries each.
// Q [B,H,NQ,64], K [B,H,NK,64], V^T [B,H,64,NK]; out bf16 [B,NQ,DO].
// ---------------------------------------------------------------------------
__global__ __launch_bounds__(128) void attn_kernel(
    const u16* __restrict__ qh, const u16* __restrict__ kh,
    const u16* __restrict__ vt, u16* __restrict__ o16) {
  __shared__ __align__(16) u16 ldsP[4][16][32];
  const int wave = threadIdx.x >> 5, lane = threadIdx.x & 31;
  const int b = blockIdx.z, h = blockIdx.y;
  const int i0 = blockIdx.x * 64 + wave * 16;
  const int lrow  = lane & 15;
  const int klo   = (lane < 16) ? 0 : 8;
  const int mhalf = (lane >> 4) << 3;

  const u16* qb = qh + (((long)b * Hc + h) * NQc) * HDc;
  const u16* kb = kh + (((long)b * Hc + h) * NKc) * HDc;
  const u16* vb = vt + (((long)b * Hc + h) * HDc) * NKc;

  Frag16 aq[2];
  #pragma unroll
  for (int kk = 0; kk < 2; ++kk) {
    const u16* p = qb + (long)(i0 + lrow) * HDc + kk * 32;
    aq[kk].q[0] = *(const uint4*)(p + klo);
    aq[kk].q[1] = *(const uint4*)(p + 16 + klo);
  }

  Acc8 acc[4];
  #pragma unroll
  for (int t = 0; t < 4; ++t)
    #pragma unroll
    for (int j = 0; j < 8; ++j) acc[t].f[j] = 0.0f;
  float mstat[8], lstat[8];
  #pragma unroll
  for (int r = 0; r < 8; ++r) { mstat[r] = -3.0e38f; lstat[r] = 0.0f; }

  const float scale = 0.125f;   // 1/sqrt(64)

  for (int jb = 0; jb < NKc / 32; ++jb) {
    // S = Q @ K^T for a 16x32 key block (2 N-tiles x 2 K-steps)
    Acc8 s[2];
    #pragma unroll
    for (int nt = 0; nt < 2; ++nt) {
      #pragma unroll
      for (int j = 0; j < 8; ++j) s[nt].f[j] = 0.0f;
      #pragma unroll
      for (int kk = 0; kk < 2; ++kk) {
        Frag16 bk;
        const u16* p = kb + (long)(jb * 32 + nt * 16 + lrow) * HDc + kk * 32;
        bk.q[0] = *(const uint4*)(p + klo);
        bk.q[1] = *(const uint4*)(p + 16 + klo);
        s[nt].v = __builtin_amdgcn_wmma_f32_16x16x32_bf16(
            false, aq[kk].v, false, bk.v, (short)0, s[nt].v, false, false);
      }
    }
    // Online softmax (row reductions across each 16-lane half)
    float pr0[8], pr1[8];
    #pragma unroll
    for (int r = 0; r < 8; ++r) {
      float s0 = s[0].f[r] * scale;
      float s1 = s[1].f[r] * scale;
      float t = fmaxf(s0, s1);
      t = fmaxf(t, __shfl_xor(t, 1, 32));
      t = fmaxf(t, __shfl_xor(t, 2, 32));
      t = fmaxf(t, __shfl_xor(t, 4, 32));
      t = fmaxf(t, __shfl_xor(t, 8, 32));
      float mn = fmaxf(mstat[r], t);
      float al = __expf(mstat[r] - mn);
      float p0 = __expf(s0 - mn);
      float p1 = __expf(s1 - mn);
      float rs = p0 + p1;
      rs += __shfl_xor(rs, 1, 32);
      rs += __shfl_xor(rs, 2, 32);
      rs += __shfl_xor(rs, 4, 32);
      rs += __shfl_xor(rs, 8, 32);
      lstat[r] = lstat[r] * al + rs;
      mstat[r] = mn;
      #pragma unroll
      for (int t2 = 0; t2 < 4; ++t2) acc[t2].f[r] *= al;
      pr0[r] = p0; pr1[r] = p1;
    }
    // C-layout -> A-layout conversion of P via per-wave LDS tile
    #pragma unroll
    for (int r = 0; r < 8; ++r) {
      ldsP[wave][mhalf + r][lrow]      = f2bf(pr0[r]);
      ldsP[wave][mhalf + r][16 + lrow] = f2bf(pr1[r]);
    }
    __syncthreads();
    Frag16 pa;
    pa.q[0] = *(const uint4*)&ldsP[wave][lrow][klo];
    pa.q[1] = *(const uint4*)&ldsP[wave][lrow][16 + klo];
    __syncthreads();
    // O += P @ V  (V^T gives contiguous B-operand columns)
    #pragma unroll
    for (int dt = 0; dt < 4; ++dt) {
      Frag16 bv;
      const u16* p = vb + (long)(dt * 16 + lrow) * NKc + jb * 32;
      bv.q[0] = *(const uint4*)(p + klo);
      bv.q[1] = *(const uint4*)(p + 16 + klo);
      acc[dt].v = __builtin_amdgcn_wmma_f32_16x16x32_bf16(
          false, pa.v, false, bv.v, (short)0, acc[dt].v, false, false);
    }
  }
  // Normalize and store bf16 o [B,NQ,DO]
  #pragma unroll
  for (int r = 0; r < 8; ++r) {
    float inv = 1.0f / lstat[r];
    int gi = i0 + mhalf + r;
    #pragma unroll
    for (int dt = 0; dt < 4; ++dt)
      o16[((long)b * NQc + gi) * DOc + h * HDc + dt * 16 + lrow] =
          f2bf(acc[dt].f[r] * inv);
  }
}

// ---------------------------------------------------------------------------
extern "C" void kernel_launch(void* const* d_in, const int* in_sizes, int n_in,
                              void* d_out, int out_size, void* d_ws, size_t ws_size,
                              hipStream_t stream) {
  const float* xk      = (const float*)d_in[0];
  const float* xq      = (const float*)d_in[1];
  const float* alpha_q = (const float*)d_in[2];
  const float* gamma_q = (const float*)d_in[3];
  const float* alpha_k = (const float*)d_in[4];
  const float* gamma_k = (const float*)d_in[5];
  const float* Wq      = (const float*)d_in[6];
  const float* Wkv     = (const float*)d_in[7];
  const float* Wproj   = (const float*)d_in[8];
  const float* bproj   = (const float*)d_in[9];
  const float* alpha2  = (const float*)d_in[10];
  const float* gamma2  = (const float*)d_in[11];
  const float* W1      = (const float*)d_in[12];
  const float* b1      = (const float*)d_in[13];
  const float* W2      = (const float*)d_in[14];
  const float* b2      = (const float*)d_in[15];
  float* out = (float*)d_out;

  char* ws = (char*)d_ws;
  size_t off = 0;
  auto alloc = [&](size_t bytes) -> char* {
    off = (off + 255) & ~(size_t)255;
    char* p = ws + off;
    off += bytes;
    return p;
  };
  const size_t ROWS = (size_t)Bc * NQc;                 // 4096
  u16*   xq_n   = (u16*)  alloc(ROWS * DOc * 2);
  u16*   xk_n   = (u16*)  alloc(ROWS * DOc * 2);
  u16*   WqT    = (u16*)  alloc((size_t)DOc * DOc * 2);
  u16*   WkvT   = (u16*)  alloc((size_t)2 * DOc * DOc * 2);
  u16*   WprojT = (u16*)  alloc((size_t)DOc * DOc * 2);
  u16*   W1T    = (u16*)  alloc((size_t)HIDc * DOc * 2);
  u16*   W2T    = (u16*)  alloc((size_t)DOc * HIDc * 2);
  u16*   q_h    = (u16*)  alloc(ROWS * DOc * 2);
  u16*   k_h    = (u16*)  alloc(ROWS * DOc * 2);
  u16*   v_t    = (u16*)  alloc(ROWS * DOc * 2);
  u16*   o16    = (u16*)  alloc(ROWS * DOc * 2);
  float* xf     = (float*)alloc(ROWS * DOc * 4);
  u16*   h0     = (u16*)  alloc(ROWS * DOc * 2);
  u16*   h1     = (u16*)  alloc(ROWS * HIDc * 2);
  (void)ws_size; (void)in_sizes; (void)n_in; (void)out_size;

  const long total = (long)ROWS * DOc;
  dyt_cast_kernel<<<(int)(total / 256), 256, 0, stream>>>(xq, alpha_q, gamma_q, xq_n, DOc, total);
  dyt_cast_kernel<<<(int)(total / 256), 256, 0, stream>>>(xk, alpha_k, gamma_k, xk_n, DOc, total);

  dim3 tb(32, 8);
  transpose_cast_kernel<<<dim3(DOc / 32,  DOc / 32),  tb, 0, stream>>>(Wq,    WqT,    DOc,  DOc);
  transpose_cast_kernel<<<dim3(HIDc / 32, DOc / 32),  tb, 0, stream>>>(Wkv,   WkvT,   DOc,  2 * DOc);
  transpose_cast_kernel<<<dim3(DOc / 32,  DOc / 32),  tb, 0, stream>>>(Wproj, WprojT, DOc,  DOc);
  transpose_cast_kernel<<<dim3(HIDc / 32, DOc / 32),  tb, 0, stream>>>(W1,    W1T,    DOc,  HIDc);
  transpose_cast_kernel<<<dim3(DOc / 32,  HIDc / 32), tb, 0, stream>>>(W2,    W2T,    HIDc, DOc);

  // Q projection -> per-head layout
  gemm_wmma_kernel<0><<<dim3(ROWS / 128, DOc / 128), 256, 0, stream>>>(
      xq_n, WqT, (int)ROWS, DOc, DOc, nullptr, nullptr, nullptr, nullptr,
      q_h, nullptr, nullptr);
  // KV projection -> K per-head + V transposed
  gemm_wmma_kernel<1><<<dim3(ROWS / 128, (2 * DOc) / 128), 256, 0, stream>>>(
      xk_n, WkvT, (int)ROWS, 2 * DOc, DOc, nullptr, nullptr, nullptr, nullptr,
      k_h, v_t, nullptr);
  // Flash attention
  attn_kernel<<<dim3(NQc / 64, Hc, Bc), 128, 0, stream>>>(q_h, k_h, v_t, o16);
  // Output projection: x = o@Wproj + bproj; h0 = bf16(dyt(x))
  gemm_wmma_kernel<2><<<dim3(ROWS / 128, DOc / 128), 256, 0, stream>>>(
      o16, WprojT, (int)ROWS, DOc, DOc, bproj, gamma2, alpha2, nullptr,
      h0, nullptr, xf);
  // FFN up: h1 = gelu(h0@W1 + b1)
  gemm_wmma_kernel<3><<<dim3(ROWS / 128, HIDc / 128), 256, 0, stream>>>(
      h0, W1T, (int)ROWS, HIDc, DOc, b1, nullptr, nullptr, nullptr,
      h1, nullptr, nullptr);
  // FFN down + residual: out = x + h1@W2 + b2
  gemm_wmma_kernel<4><<<dim3(ROWS / 128, DOc / 128), 256, 0, stream>>>(
      h1, W2T, (int)ROWS, DOc, HIDc, b2, nullptr, nullptr, xf,
      nullptr, nullptr, out);
}
